// PointNet2_44693429682245
// MI455X (gfx1250) — compile-verified, matrix operand order checked by reading, not run on hardware
//
#include <hip/hip_runtime.h>
#include <hip/hip_fp16.h>
#include <stdint.h>

#define FLTMAX 3.402823466e+38f

typedef __attribute__((ext_vector_type(16))) _Float16 v16h;
typedef __attribute__((ext_vector_type(8)))  _Float16 v8h;
typedef __attribute__((ext_vector_type(8)))  float    v8f;

#define DEVI __device__ __forceinline__

DEVI v8f vzero() {
  v8f z;
#pragma unroll
  for (int i = 0; i < 8; ++i) z[i] = 0.f;
  return z;
}

DEVI v8f wmma_f32_f16(v16h a, v16h b, v8f c) {
  // D = A(16x32 f16) * B(32x16 f16) + C(16x16 f32)
  return __builtin_amdgcn_wmma_f32_16x16x32_f16(false, a, false, b, (short)0, c, false, false);
}

// A-tile staged row-major in LDS as [16][KPAD] f16.
// CDNA5 16-bit A layout: lane<16  -> M=lane,    K = {8g..8g+7} in e0..7, {16+8g..} in e8..15 (g=0)
//                        lane>=16 -> M=lane-16, same with g=1.
DEVI v16h load_afrag(const _Float16* At, int kpad, int kt, int lane) {
  const int g = (lane >> 4) & 1;
  const int M = lane & 15;
  const _Float16* row = At + M * kpad + kt * 32;
  v8h lo = *(const v8h*)(row + 8 * g);
  v8h hi = *(const v8h*)(row + 16 + 8 * g);
  v16h a;
#pragma unroll
  for (int i = 0; i < 8; ++i) { a[i] = lo[i]; a[8 + i] = hi[i]; }
  return a;
}

// B fragment built from a row-major f32 weight matrix W[rows][cols] (K x N),
// assuming B mirrors A's K mapping with N striped across lanes.
DEVI v16h load_bfrag(const float* W, int rows, int cols, int kt, int nt, int lane) {
  const int g = (lane >> 4) & 1;
  const int n = (lane & 15) + nt * 16;
  v16h b;
#pragma unroll
  for (int e = 0; e < 16; ++e) {
    const int k = kt * 32 + ((e < 8) ? (e + 8 * g) : (16 + (e - 8) + 8 * g));
    float w = 0.f;
    if (k < rows && n < cols) w = W[k * cols + n];
    b[e] = (_Float16)w;
  }
  return b;
}

// ---------------------------------------------------------------------------
// Farthest point sampling: one block per batch, distances register-resident.
// Latency-bound serial chain -> minimize per-step critical path.
// ---------------------------------------------------------------------------
template<int BLOCK, int PTS>
__global__ void __launch_bounds__(BLOCK)
fps_kernel(const float* __restrict__ pos, int* __restrict__ idx_out,
           float* __restrict__ posq_out, int m) {
  constexpr int N  = BLOCK * PTS;
  constexpr int NW = BLOCK / 32;
  const int b = blockIdx.x;
  const float* P = pos + (size_t)b * N * 3;
  const int tid = threadIdx.x, lane = tid & 31, wv = tid >> 5;
  __shared__ float sval[NW];
  __shared__ int   sidx[NW];
  __shared__ int   snext;
  float px[PTS], py[PTS], pz[PTS], dist[PTS];
#pragma unroll
  for (int i = 0; i < PTS; ++i) {
    const int p = i * BLOCK + tid;
    px[i] = P[p * 3 + 0]; py[i] = P[p * 3 + 1]; pz[i] = P[p * 3 + 2];
    dist[i] = FLTMAX;
  }
  float lx = P[0], ly = P[1], lz = P[2];
  if (tid == 0) {
    idx_out[(size_t)b * m] = 0;
    posq_out[((size_t)b * m) * 3 + 0] = lx;
    posq_out[((size_t)b * m) * 3 + 1] = ly;
    posq_out[((size_t)b * m) * 3 + 2] = lz;
  }
  for (int s = 1; s < m; ++s) {
    float best = -FLTMAX; int bidx = 0;
#pragma unroll
    for (int i = 0; i < PTS; ++i) {
      const float dx = px[i] - lx, dy = py[i] - ly, dz = pz[i] - lz;
      float d = dx * dx + dy * dy + dz * dz;
      d = fminf(dist[i], d);
      dist[i] = d;
      if (d > best) { best = d; bidx = i * BLOCK + tid; }
    }
#pragma unroll
    for (int off = 16; off; off >>= 1) {
      const float ov = __shfl_xor(best, off);
      const int   oi = __shfl_xor(bidx, off);
      if (ov > best || (ov == best && oi < bidx)) { best = ov; bidx = oi; }
    }
    if (lane == 0) { sval[wv] = best; sidx[wv] = bidx; }
    __syncthreads();
    if (wv == 0) {
      float v = (lane < NW) ? sval[lane] : -FLTMAX;
      int   ix = (lane < NW) ? sidx[lane] : 0;
#pragma unroll
      for (int off = 16; off; off >>= 1) {
        const float ov = __shfl_xor(v, off);
        const int   oi = __shfl_xor(ix, off);
        if (ov > v || (ov == v && oi < ix)) { v = ov; ix = oi; }
      }
      if (lane == 0) { snext = ix; idx_out[(size_t)b * m + s] = ix; }
    }
    __syncthreads();
    const int nxt = snext;
    lx = P[(size_t)nxt * 3 + 0]; ly = P[(size_t)nxt * 3 + 1]; lz = P[(size_t)nxt * 3 + 2];
    if (tid == 0) {
      posq_out[((size_t)b * m + s) * 3 + 0] = lx;
      posq_out[((size_t)b * m + s) * 3 + 1] = ly;
      posq_out[((size_t)b * m + s) * 3 + 2] = lz;
    }
    __syncthreads();
  }
}

// ---------------------------------------------------------------------------
// K nearest within radius: LDS-tiled brute force, register top-K (replace-max),
// next tile prefetched (global_prefetch_b8) while current tile is consumed.
// Invalid (outside radius) -> index -1.
// ---------------------------------------------------------------------------
template<int K>
__global__ void radius_kernel(const float* __restrict__ pos_src,
                              const float* __restrict__ posq,
                              int* __restrict__ nbr, int n, int m, float r2) {
  constexpr int TILE = 256;
  __shared__ float sp[TILE * 3];
  const int b = blockIdx.y;
  const int q = blockIdx.x * blockDim.x + threadIdx.x;
  float qx = 0.f, qy = 0.f, qz = 0.f;
  if (q < m) {
    const float* pq = posq + ((size_t)b * m + q) * 3;
    qx = pq[0]; qy = pq[1]; qz = pq[2];
  }
  float dd[K]; int ii[K];
#pragma unroll
  for (int j = 0; j < K; ++j) { dd[j] = FLTMAX; ii[j] = -1; }
  float kmax = FLTMAX; int kslot = 0;
  for (int t0 = 0; t0 < n; t0 += TILE) {
    const int cnt = (TILE < n - t0) ? TILE : (n - t0);
    for (int i = threadIdx.x; i < cnt * 3; i += blockDim.x)
      sp[i] = pos_src[((size_t)b * n + t0) * 3 + i];
    // Prefetch next tile into cache while this one is consumed.
    if (t0 + TILE < n) {
      const float* np = pos_src + ((size_t)b * n + t0 + TILE) * 3;
      __builtin_prefetch(np + (size_t)threadIdx.x * 3, 0, 1);
    }
    __syncthreads();
    if (q < m) {
      for (int s = 0; s < cnt; ++s) {
        const float dx = sp[s * 3 + 0] - qx;
        const float dy = sp[s * 3 + 1] - qy;
        const float dz = sp[s * 3 + 2] - qz;
        const float d = dx * dx + dy * dy + dz * dz;
        if (d < kmax) {
#pragma unroll
          for (int j = 0; j < K; ++j)
            if (j == kslot) { dd[j] = d; ii[j] = t0 + s; }
          float km = -FLTMAX; int ks = 0;
#pragma unroll
          for (int j = 0; j < K; ++j)
            if (dd[j] > km) { km = dd[j]; ks = j; }
          kmax = km; kslot = ks;
        }
      }
    }
    __syncthreads();
  }
  if (q < m) {
#pragma unroll
    for (int j = 0; j < K; ++j)
      nbr[((size_t)b * m + q) * K + j] = (dd[j] <= r2) ? ii[j] : -1;
  }
}

// ---------------------------------------------------------------------------
// Pass 1: gather msg = [x_j, pos_j - pos_i] (or [x, pos] for SA4), f16 A tile
// in LDS, WMMA with W1, +b1, write h (f32), accumulate masked sum/sumsq/count.
// One 16-row tile per wave. Stats epilogue is branchless (cndmask+fma, no
// exec-mask churn after the WMMA hazard window).
// ---------------------------------------------------------------------------
template<int K, int CINX, int CH, int KT, int NT, bool ABS_POS>
__global__ void __launch_bounds__(256)
pass1_kernel(const float* __restrict__ xsrc, const float* __restrict__ possrc,
             const float* __restrict__ posq, const int* __restrict__ nbr,
             const float* __restrict__ W1, const float* __restrict__ b1v,
             float* __restrict__ hout, float* __restrict__ stats,
             int* __restrict__ cnt, int nsrc, int m, int ntiles) {
  constexpr int CIN   = CINX + 3;
  constexpr int KPAD  = KT * 32;
  constexpr int CHPAD = NT * 16;
  constexpr int WAVES = 8;
  __shared__ __align__(16) _Float16 A[WAVES][16][KPAD];
  __shared__ float rS[WAVES], rS2[WAVES];
  __shared__ int   rC[WAVES];
  const int lane = threadIdx.x & 31, wv = threadIdx.x >> 5;
  const int t = blockIdx.x * WAVES + wv;
  float s = 0.f, s2 = 0.f; int c = 0;
  if (t < ntiles) {
    const int row0 = t * 16;
    const int b = row0 / (m * K);
    const int rem = row0 - b * m * K;
    const int q = rem / K;
    const int k0 = rem - q * K;
    bool valid = false;
    if (lane < 16) {
      const int j = nbr ? nbr[((size_t)(b * m + q)) * K + k0 + lane]
                        : (q * K + k0 + lane);
      valid = (j >= 0);
      const int jj = valid ? j : 0;
      _Float16* row = &A[wv][lane][0];
      const float* xs = xsrc + ((size_t)b * nsrc + jj) * CINX;
#pragma unroll
      for (int cc = 0; cc < CINX; ++cc)
        row[cc] = valid ? (_Float16)xs[cc] : (_Float16)0.f;
      const float* ps = possrc + ((size_t)b * nsrc + jj) * 3;
      float qx = 0.f, qy = 0.f, qz = 0.f;
      if constexpr (!ABS_POS) {
        const float* pq = posq + ((size_t)b * m + q) * 3;
        qx = pq[0]; qy = pq[1]; qz = pq[2];
      }
      row[CINX + 0] = valid ? (_Float16)(ps[0] - qx) : (_Float16)0.f;
      row[CINX + 1] = valid ? (_Float16)(ps[1] - qy) : (_Float16)0.f;
      row[CINX + 2] = valid ? (_Float16)(ps[2] - qz) : (_Float16)0.f;
#pragma unroll
      for (int cc = CIN; cc < KPAD; ++cc) row[cc] = (_Float16)0.f;
    }
    const unsigned maskbits = (unsigned)(__ballot(valid) & 0xFFFFull);
    __builtin_amdgcn_wave_barrier();

    v16h bf[KT][NT];
#pragma unroll
    for (int kt = 0; kt < KT; ++kt)
#pragma unroll
      for (int nt = 0; nt < NT; ++nt)
        bf[kt][nt] = load_bfrag(W1, CIN, CH, kt, nt, lane);

    v8f acc[NT];
#pragma unroll
    for (int nt = 0; nt < NT; ++nt) acc[nt] = vzero();
#pragma unroll
    for (int kt = 0; kt < KT; ++kt) {
      const v16h a = load_afrag(&A[wv][0][0], KPAD, kt, lane);
#pragma unroll
      for (int nt = 0; nt < NT; ++nt) acc[nt] = wmma_f32_f16(a, bf[kt][nt], acc[nt]);
    }
    const int hi = lane >> 4, nn = lane & 15;
#pragma unroll
    for (int nt = 0; nt < NT; ++nt) {
      const int ncol = nn + nt * 16;
      const bool col_ok = (ncol < CH);
      const float bval = b1v[col_ok ? ncol : (CH - 1)];  // unguarded load, select after
      const float badd = col_ok ? bval : 0.f;
#pragma unroll
      for (int v = 0; v < 8; ++v) {
        const int M = v + 8 * hi;
        const float val = acc[nt][v] + badd;
        hout[((size_t)row0 + M) * CHPAD + ncol] = val;
        // branchless masked accumulation
        const float w = (col_ok && ((maskbits >> M) & 1u)) ? 1.f : 0.f;
        const float wv_ = w * val;
        s  = fmaf(w,   val, s);
        s2 = fmaf(wv_, val, s2);
      }
    }
    c = (lane == 0) ? (int)__popc(maskbits) : 0;
  }
#pragma unroll
  for (int off = 16; off; off >>= 1) {
    s  += __shfl_xor(s,  off);
    s2 += __shfl_xor(s2, off);
    c  += __shfl_xor(c,  off);
  }
  if (lane == 0) { rS[wv] = s; rS2[wv] = s2; rC[wv] = c; }
  __syncthreads();
  if (threadIdx.x == 0) {
    float S = 0.f, S2 = 0.f; int C = 0;
#pragma unroll
    for (int w = 0; w < WAVES; ++w) { S += rS[w]; S2 += rS2[w]; C += rC[w]; }
    atomicAdd(&stats[0], S);
    atomicAdd(&stats[1], S2);
    atomicAdd(cnt, C);
  }
}

__global__ void finalize_kernel(float* stats, const int* cnt, int ch) {
  if (threadIdx.x == 0 && blockIdx.x == 0) {
    const float n = (float)(*cnt) * (float)ch;
    const float mean = stats[0] / n;
    const float var = stats[1] / n - mean * mean;
    const float sd = sqrtf(fmaxf(var, 0.f));
    stats[2] = mean;
    stats[3] = 1.f / (sd + 1e-5f);
  }
}

// ---------------------------------------------------------------------------
// Pass 2: normalize+ReLU -> f16 A tile, WMMA with W2, masked max over the K
// rows of each query (branchless per-lane max of its D rows, cross-half shfl),
// +b2. One query group (K rows = K/16 tiles) per wave.
// ---------------------------------------------------------------------------
template<int K, int CH, int CHPAD, int K2T, int COUT, int N2T>
__global__ void __launch_bounds__(256)
pass2_kernel(const float* __restrict__ hin, const int* __restrict__ nbr,
             const float* __restrict__ stats, const float* __restrict__ gvec,
             const float* __restrict__ bevec, const float* __restrict__ W2,
             const float* __restrict__ b2v, float* __restrict__ xout, int ngroups) {
  constexpr int K2PAD = K2T * 32;
  constexpr int GT    = K / 16;
  constexpr int WAVES = 8;
  __shared__ __align__(16) _Float16 A[WAVES][16][K2PAD];
  const int lane = threadIdx.x & 31, wv = threadIdx.x >> 5;
  const int gidx = blockIdx.x * WAVES + wv;
  if (gidx >= ngroups) return;
  const float mean = stats[2], rstd = stats[3];
  v16h bf[K2T][N2T];
#pragma unroll
  for (int kt = 0; kt < K2T; ++kt)
#pragma unroll
    for (int nt = 0; nt < N2T; ++nt)
      bf[kt][nt] = load_bfrag(W2, CH, COUT, kt, nt, lane);
  float cmax[N2T];
#pragma unroll
  for (int nt = 0; nt < N2T; ++nt) cmax[nt] = -FLTMAX;
  const int rowb = gidx * K;
#pragma unroll
  for (int gt = 0; gt < GT; ++gt) {
    const int row0 = rowb + gt * 16;
    bool valid = false;
    if (lane < 16) valid = nbr ? (nbr[(size_t)row0 + lane] >= 0) : true;
    const unsigned maskbits = (unsigned)(__ballot(valid) & 0xFFFFull);
    for (int idx = lane; idx < 16 * K2PAD; idx += 32) {
      const int rr = idx / K2PAD, cc = idx - rr * K2PAD;
      float v = 0.f;
      if (cc < CH) {
        v = hin[((size_t)row0 + rr) * CHPAD + cc];
        v = (v - mean) * rstd * gvec[cc] + bevec[cc];
        v = fmaxf(v, 0.f);
      }
      A[wv][rr][cc] = (_Float16)v;
    }
    __builtin_amdgcn_wave_barrier();
    v8f acc[N2T];
#pragma unroll
    for (int nt = 0; nt < N2T; ++nt) acc[nt] = vzero();
#pragma unroll
    for (int kt = 0; kt < K2T; ++kt) {
      const v16h a = load_afrag(&A[wv][0][0], K2PAD, kt, lane);
#pragma unroll
      for (int nt = 0; nt < N2T; ++nt) acc[nt] = wmma_f32_f16(a, bf[kt][nt], acc[nt]);
    }
    const int hi = lane >> 4;
#pragma unroll
    for (int nt = 0; nt < N2T; ++nt)
#pragma unroll
      for (int v = 0; v < 8; ++v) {
        const int M = v + 8 * hi;
        const float vv = ((maskbits >> M) & 1u) ? acc[nt][v] : -FLTMAX;  // branchless
        cmax[nt] = fmaxf(cmax[nt], vv);
      }
    __builtin_amdgcn_wave_barrier();
  }
#pragma unroll
  for (int nt = 0; nt < N2T; ++nt)
    cmax[nt] = fmaxf(cmax[nt], __shfl_xor(cmax[nt], 16));
  if (lane < 16) {
#pragma unroll
    for (int nt = 0; nt < N2T; ++nt) {
      const int ncol = lane + nt * 16;
      if (ncol < COUT) xout[(size_t)gidx * COUT + ncol] = cmax[nt] + b2v[ncol];
    }
  }
}

// ---------------------------------------------------------------------------
// Head: max over 8 group-rows per batch, Linear(64,64), Linear(64,1).
// ---------------------------------------------------------------------------
__global__ void head_kernel(const float* __restrict__ g4,
                            const float* __restrict__ W1, const float* __restrict__ b1,
                            const float* __restrict__ W2, const float* __restrict__ b2,
                            float* __restrict__ out) {
  const int b = blockIdx.x;
  const int n = threadIdx.x;  // 64 threads
  __shared__ float gm[64], y1[64], red[64];
  float mx = -FLTMAX;
#pragma unroll
  for (int t = 0; t < 8; ++t) mx = fmaxf(mx, g4[((size_t)b * 8 + t) * 64 + n]);
  gm[n] = mx;
  __syncthreads();
  float acc = b1[n];
  for (int k = 0; k < 64; ++k) acc += gm[k] * W1[k * 64 + n];
  y1[n] = acc;
  __syncthreads();
  red[n] = y1[n] * W2[n];
  __syncthreads();
  for (int s = 32; s; s >>= 1) {
    if (n < s) red[n] += red[n + s];
    __syncthreads();
  }
  if (n == 0) out[b] = red[0] + b2[0];
}

// ---------------------------------------------------------------------------
extern "C" void kernel_launch(void* const* d_in, const int* in_sizes, int n_in,
                              void* d_out, int out_size, void* d_ws, size_t ws_size,
                              hipStream_t stream) {
  (void)in_sizes; (void)n_in; (void)out_size; (void)ws_size;
  // Inputs (jax pytree: dict keys sorted): x, pos, batch, params{head,sa1..sa4}
  const float* x    = (const float*)d_in[0];   // [8,8192,6]
  const float* pos  = (const float*)d_in[1];   // [8,8192,3]
  const float* hW1  = (const float*)d_in[3];   // head: W1,W2,b1,b2
  const float* hW2  = (const float*)d_in[4];
  const float* hb1  = (const float*)d_in[5];
  const float* hb2  = (const float*)d_in[6];
  // per-SA order: W1, W2, b1, b2, be, g
  const float* s1W1 = (const float*)d_in[7];
  const float* s1W2 = (const float*)d_in[8];
  const float* s1b1 = (const float*)d_in[9];
  const float* s1b2 = (const float*)d_in[10];
  const float* s1be = (const float*)d_in[11];
  const float* s1g  = (const float*)d_in[12];
  const float* s2W1 = (const float*)d_in[13];
  const float* s2W2 = (const float*)d_in[14];
  const float* s2b1 = (const float*)d_in[15];
  const float* s2b2 = (const float*)d_in[16];
  const float* s2be = (const float*)d_in[17];
  const float* s2g  = (const float*)d_in[18];
  const float* s3W1 = (const float*)d_in[19];
  const float* s3W2 = (const float*)d_in[20];
  const float* s3b1 = (const float*)d_in[21];
  const float* s3b2 = (const float*)d_in[22];
  const float* s3be = (const float*)d_in[23];
  const float* s3g  = (const float*)d_in[24];
  const float* s4W1 = (const float*)d_in[25];
  const float* s4W2 = (const float*)d_in[26];
  const float* s4b1 = (const float*)d_in[27];
  const float* s4b2 = (const float*)d_in[28];
  const float* s4be = (const float*)d_in[29];
  const float* s4g  = (const float*)d_in[30];
  float* out = (float*)d_out;

  uint8_t* base = (uint8_t*)d_ws;
  size_t off = 0;
  auto alloc = [&](size_t bytes) -> void* {
    void* p = base + off;
    off = (off + bytes + 255) & ~(size_t)255;
    return p;
  };
  float* stats = (float*)alloc(16 * sizeof(float) + 4 * sizeof(int));
  int* cnts = (int*)(stats + 16);
  int*   idx1  = (int*)  alloc((size_t)8 * 2048 * 4);
  float* posq1 = (float*)alloc((size_t)8 * 2048 * 3 * 4);
  int*   nbr1  = (int*)  alloc((size_t)8 * 2048 * 16 * 4);
  float* h1    = (float*)alloc((size_t)8 * 2048 * 16 * 16 * 4);
  float* x1    = (float*)alloc((size_t)8 * 2048 * 8 * 4);
  int*   idx2  = (int*)  alloc((size_t)8 * 512 * 4);
  float* posq2 = (float*)alloc((size_t)8 * 512 * 3 * 4);
  int*   nbr2  = (int*)  alloc((size_t)8 * 512 * 32 * 4);
  float* h2    = (float*)alloc((size_t)8 * 512 * 32 * 16 * 4);
  float* x2    = (float*)alloc((size_t)8 * 512 * 16 * 4);
  int*   idx3  = (int*)  alloc((size_t)8 * 128 * 4);
  float* posq3 = (float*)alloc((size_t)8 * 128 * 3 * 4);
  int*   nbr3  = (int*)  alloc((size_t)8 * 128 * 64 * 4);
  float* h3    = (float*)alloc((size_t)8 * 128 * 64 * 32 * 4);
  float* x3    = (float*)alloc((size_t)8 * 128 * 32 * 4);
  float* h4    = (float*)alloc((size_t)1024 * 48 * 4);
  float* g4    = (float*)alloc((size_t)64 * 64 * 4);

  hipMemsetAsync(stats, 0, 16 * sizeof(float) + 4 * sizeof(int), stream);

  // ---------------- SA1: n=8192 -> m=2048, r=0.1, K=16, MLP [9,8,8]
  fps_kernel<1024, 8><<<8, 1024, 0, stream>>>(pos, idx1, posq1, 2048);
  radius_kernel<16><<<dim3(8, 8), 256, 0, stream>>>(pos, posq1, nbr1, 8192, 2048, 0.01f);
  pass1_kernel<16, 6, 8, 1, 1, false><<<2048, 256, 0, stream>>>(
      x, pos, posq1, nbr1, s1W1, s1b1, h1, stats + 0, cnts + 0, 8192, 2048, 16384);
  finalize_kernel<<<1, 1, 0, stream>>>(stats + 0, cnts + 0, 8);
  pass2_kernel<16, 8, 16, 1, 8, 1><<<2048, 256, 0, stream>>>(
      h1, nbr1, stats + 0, s1g, s1be, s1W2, s1b2, x1, 16384);

  // ---------------- SA2: n=2048 -> m=512, r=0.2, K=32, MLP [11,12,16]
  fps_kernel<1024, 2><<<8, 1024, 0, stream>>>(posq1, idx2, posq2, 512);
  radius_kernel<32><<<dim3(2, 8), 256, 0, stream>>>(posq1, posq2, nbr2, 2048, 512, 0.04f);
  pass1_kernel<32, 8, 12, 1, 1, false><<<1024, 256, 0, stream>>>(
      x1, posq1, posq2, nbr2, s2W1, s2b1, h2, stats + 4, cnts + 1, 2048, 512, 8192);
  finalize_kernel<<<1, 1, 0, stream>>>(stats + 4, cnts + 1, 12);
  pass2_kernel<32, 12, 16, 1, 16, 1><<<512, 256, 0, stream>>>(
      h2, nbr2, stats + 4, s2g, s2be, s2W2, s2b2, x2, 4096);

  // ---------------- SA3: n=512 -> m=128, r=0.4, K=64, MLP [19,24,32]
  fps_kernel<512, 1><<<8, 512, 0, stream>>>(posq2, idx3, posq3, 128);
  radius_kernel<64><<<dim3(1, 8), 128, 0, stream>>>(posq2, posq3, nbr3, 512, 128, 0.16f);
  pass1_kernel<64, 16, 24, 1, 2, false><<<512, 256, 0, stream>>>(
      x2, posq2, posq3, nbr3, s3W1, s3b1, h3, stats + 8, cnts + 2, 512, 128, 4096);
  finalize_kernel<<<1, 1, 0, stream>>>(stats + 8, cnts + 2, 24);
  pass2_kernel<64, 24, 32, 1, 32, 2><<<128, 256, 0, stream>>>(
      h3, nbr3, stats + 8, s3g, s3be, s3W2, s3b2, x3, 1024);

  // ---------------- SA4 (global): rows [8][128] = [8][8][16], MLP [35,48,64]
  pass1_kernel<16, 32, 48, 2, 3, true><<<8, 256, 0, stream>>>(
      x3, posq3, nullptr, nullptr, s4W1, s4b1, h4, stats + 12, cnts + 3, 128, 8, 64);
  finalize_kernel<<<1, 1, 0, stream>>>(stats + 12, cnts + 3, 48);
  pass2_kernel<16, 48, 48, 2, 64, 4><<<8, 256, 0, stream>>>(
      h4, nullptr, stats + 12, s4g, s4be, s4W2, s4b2, g4, 64);

  // ---------------- Head
  head_kernel<<<8, 64, 0, stream>>>(g4, hW1, hb1, hW2, hb2, out);
}
